// DiffNet_SocialGCN_12025908429033
// MI455X (gfx1250) — matrix operand order repaired and checked by reading, not compiled, not run on hardware
//
#include <hip/hip_runtime.h>

#define NN 16384
#define HH 64
#define KSPLIT 8
#define GRAN 4                              // K-chunks staged per async group
#define CHUNKS (NN / 32 / KSPLIT)           // 64 chunks of K=32 per split
#define NGROUP (CHUNKS / GRAN)              // 16 double-buffered groups
#define CHUNK_BYTES 4096                    // 4 t-tiles * 32 lanes * 16 f16 * 2B

typedef __attribute__((ext_vector_type(16))) _Float16 v16h;
typedef __attribute__((ext_vector_type(8)))  float    v8f;
typedef __attribute__((ext_vector_type(4)))  float    f32x4;

// Flat (generic) pointer -> 32-bit LDS byte offset (addrspacecast truncates to
// the wave's LDS window), as required by GLOBAL_LOAD_ASYNC_TO_LDS's VDST.
__device__ __forceinline__ unsigned lds_off(const void* p) {
  return (unsigned)(unsigned long long)(__attribute__((address_space(3))) const char*)p;
}

// ---------------------------------------------------------------------------
// Pack x[N][64] (f32, row-major) into a contiguous-k WMMA B-operand layout as f16 hi/lo.
// B element (K=kr, N=nr) of chunk kg, N-tile t:
//   lane = nr + (kr>=16 ? 16 : 0), slot s = kr&15  (16 f16/lane = 8 VGPRs)
//   offset = ((kg*4 + t)*32 + lane)*16 + s   (f16 units)
// ---------------------------------------------------------------------------
__global__ __launch_bounds__(256) void pack_x_f16(const float* __restrict__ x,
                                                  _Float16* __restrict__ hi,
                                                  _Float16* __restrict__ lo) {
  int idx = blockIdx.x * 256 + threadIdx.x;   // 0 .. N*H-1
  int r = idx >> 6;                           // K index (row of x)
  int n = idx & 63;                           // N index (col of x)
  float a = x[idx];
  _Float16 h = (_Float16)a;
  _Float16 l = (_Float16)(a - (float)h);
  int kg = r >> 5, kr = r & 31;
  int t  = n >> 4, nr = n & 15;
  int lane = nr + (kr & 16);
  int s = kr & 15;
  int o = (((kg * 4 + t) * 32 + lane) << 4) + s;
  hi[o] = h;
  lo[o] = l;
}

// ---------------------------------------------------------------------------
// Core GEMM: part[kz] = adj[:, kz-range] @ x[kz-range, :]
// Grid (N/256, KSPLIT), block 256 = 8 waves; wave owns 32 rows x 64 cols.
// fp32 emulated via f16 hi/lo: acc += ahi*bhi + alo*bhi + ahi*blo.
// B is staged per-workgroup into LDS with async global->LDS copies
// (double-buffered groups of GRAN chunks), waves consume via ds_load_b128.
// ---------------------------------------------------------------------------
__global__ __launch_bounds__(256) void gcn_agg_wmma(const float* __restrict__ adj,
                                                    const _Float16* __restrict__ bhi,
                                                    const _Float16* __restrict__ blo,
                                                    float* __restrict__ part) {
  __shared__ _Float16 sHi[2][GRAN * CHUNK_BYTES / 2];   // 2 x 16 KB
  __shared__ _Float16 sLo[2][GRAN * CHUNK_BYTES / 2];   // 2 x 16 KB

  const int tid   = threadIdx.x;
  const int wave  = tid >> 5;
  const int lane  = tid & 31;
  const int lhalf = lane >> 4;
  const int lrow  = lane & 15;
  const int rowBase = blockIdx.x * 256 + wave * 32;
  const int kc0     = blockIdx.y * CHUNKS;

  v8f acc[2][4] = {};

  // A layout (contiguous K order used here, matching the B pack): lane<16 -> row lrow, K {0..15};
  // lane>=16 -> row lrow, K {16..31}.
  const float* a0 = adj + (size_t)(rowBase + lrow) * NN + (size_t)kc0 * 32 + lhalf * 16;
  const float* a1 = a0 + (size_t)16 * NN;

  const char* gHi = (const char*)bhi + (size_t)kc0 * (CHUNK_BYTES / 32);  // kc0*128 f16 *... = kc0*... 
  const char* gLo = (const char*)blo + (size_t)kc0 * (CHUNK_BYTES / 32);
  // NOTE: per chunk the packed stride is CHUNK_BYTES bytes; element index of
  // chunk kc starts at kc*2048 f16 = kc*4096 bytes.
  gHi = (const char*)bhi + (size_t)kc0 * CHUNK_BYTES;
  gLo = (const char*)blo + (size_t)kc0 * CHUNK_BYTES;

  // Issue async global->LDS copies of GRAN chunks (hi+lo) into buffer `buf`.
  // Per wave: 8 async instructions (ASYNCcnt += 8), completing in order.
  auto stage = [&](int grp, int buf) {
    const char* srcH = gHi + (size_t)grp * GRAN * CHUNK_BYTES;
    const char* srcL = gLo + (size_t)grp * GRAN * CHUNK_BYTES;
    unsigned dH = lds_off(&sHi[buf][0]);
    unsigned dL = lds_off(&sLo[buf][0]);
#pragma unroll
    for (int i = 0; i < GRAN; ++i) {
      unsigned off = (unsigned)i * CHUNK_BYTES + (unsigned)tid * 16;
      asm volatile("global_load_async_to_lds_b128 %0, %1, off"
                   :: "v"(dH + off), "v"(srcH + off) : "memory");
      asm volatile("global_load_async_to_lds_b128 %0, %1, off"
                   :: "v"(dL + off), "v"(srcL + off) : "memory");
    }
  };

  stage(0, 0);

  for (int g = 0; g < NGROUP; ++g) {
    const int buf = g & 1;
    if (g + 1 < NGROUP) {
      stage(g + 1, buf ^ 1);
      asm volatile("s_wait_asynccnt 0x8" ::: "memory");  // retire group g only
    } else {
      asm volatile("s_wait_asynccnt 0x0" ::: "memory");
    }
    __syncthreads();   // group g visible to all waves

#pragma unroll
    for (int jj = 0; jj < GRAN; ++jj) {
      const int j = g * GRAN + jj;

      // ---- load + split A tiles (f32 -> f16 hi/lo) ----
      v16h ah[2], al[2];
#pragma unroll
      for (int tile = 0; tile < 2; ++tile) {
        const float* p = (tile ? a1 : a0) + (size_t)j * 32;
        f32x4 g0 = __builtin_nontemporal_load((const f32x4*)(p + 0));
        f32x4 g1 = __builtin_nontemporal_load((const f32x4*)(p + 4));
        f32x4 g2 = __builtin_nontemporal_load((const f32x4*)(p + 8));
        f32x4 g3 = __builtin_nontemporal_load((const f32x4*)(p + 12));
        float fs[16] = {g0.x, g0.y, g0.z, g0.w, g1.x, g1.y, g1.z, g1.w,
                        g2.x, g2.y, g2.z, g2.w, g3.x, g3.y, g3.z, g3.w};
        v16h h, l;
#pragma unroll
        for (int i = 0; i < 16; ++i) {
          _Float16 hv = (_Float16)fs[i];
          h[i] = hv;
          l[i] = (_Float16)(fs[i] - (float)hv);
        }
        ah[tile] = h;
        al[tile] = l;
      }

      __builtin_prefetch(a0 + (size_t)(j + 8) * 32, 0, 0);
      __builtin_prefetch(a1 + (size_t)(j + 8) * 32, 0, 0);

      // ---- B from LDS, 3-product f32-emulated WMMA ----
#pragma unroll
      for (int t = 0; t < 4; ++t) {
        const int be = ((jj * 4 + t) * 32 + lane) * 16;
        v16h Bh = *(const v16h*)&sHi[buf][be];
        v16h Bl = *(const v16h*)&sLo[buf][be];
#pragma unroll
        for (int tile = 0; tile < 2; ++tile) {
          acc[tile][t] = __builtin_amdgcn_wmma_f32_16x16x32_f16(
              false, ah[tile], false, Bh, (short)0, acc[tile][t], false, false);
          acc[tile][t] = __builtin_amdgcn_wmma_f32_16x16x32_f16(
              false, al[tile], false, Bh, (short)0, acc[tile][t], false, false);
          acc[tile][t] = __builtin_amdgcn_wmma_f32_16x16x32_f16(
              false, ah[tile], false, Bl, (short)0, acc[tile][t], false, false);
        }
      }
    }

    __syncthreads();   // everyone done reading buf before it is restaged
  }

  // C/D layout: VGPR i -> M = lhalf*8 + i, N = lrow (per N-tile t)
  float* outp = part + (size_t)blockIdx.y * NN * HH;
#pragma unroll
  for (int tile = 0; tile < 2; ++tile)
#pragma unroll
    for (int t = 0; t < 4; ++t)
#pragma unroll
      for (int i = 0; i < 8; ++i)
        outp[(size_t)(rowBase + tile * 16 + lhalf * 8 + i) * HH + t * 16 + lrow] =
            acc[tile][t][i];
}

// Deterministic fixed-order reduction of the K-split partials.
__global__ __launch_bounds__(256) void reduce_parts(const float* __restrict__ part,
                                                    float* __restrict__ agg) {
  size_t i = (size_t)blockIdx.x * 256 + threadIdx.x;
  float s = 0.f;
#pragma unroll
  for (int k = 0; k < KSPLIT; ++k) s += part[(size_t)k * NN * HH + i];
  agg[i] = s;
}

// out[r][n] = tanh( sum_j agg[r][j]*W[j][n] + sum_j x[r][j]*W[64+j][n] )
__global__ __launch_bounds__(256) void dense_tanh(const float* __restrict__ agg,
                                                  const float* __restrict__ x,
                                                  const float* __restrict__ W,
                                                  float* __restrict__ out) {
  __shared__ float sW[128 * 64];
  __shared__ float srow[4 * 128];
  const int t = threadIdx.x;
  for (int i = t; i < 128 * 64; i += 256) sW[i] = W[i];
  const int rbase = blockIdx.x * 4;
  for (int i = t; i < 512; i += 256) {
    int r = i >> 7, c = i & 127;
    srow[i] = (c < 64) ? agg[(size_t)(rbase + r) * HH + c]
                       : x[(size_t)(rbase + r) * HH + (c - 64)];
  }
  __syncthreads();
  const int rr = t >> 6, n = t & 63;
  const float* row = &srow[rr * 128];
  float s = 0.f;
#pragma unroll 8
  for (int j = 0; j < 128; ++j) s += row[j] * sW[j * 64 + n];
  out[(size_t)(rbase + rr) * HH + n] = tanhf(s);
}

// ---------------------------------------------------------------------------
extern "C" void kernel_launch(void* const* d_in, const int* in_sizes, int n_in,
                              void* d_out, int out_size, void* d_ws, size_t ws_size,
                              hipStream_t stream) {
  const float* x0  = (const float*)d_in[0];   // user_embs [N,64]
  const float* adj = (const float*)d_in[1];   // adj [N,N]
  const float* W   = (const float*)d_in[2];   // W [2,128,64]
  float* out = (float*)d_out;

  char* ws = (char*)d_ws;
  const size_t XB = (size_t)NN * HH * sizeof(float);       // 4 MB
  float*    part = (float*)ws;                             // KSPLIT * 4 MB
  float*    agg  = (float*)(ws + KSPLIT * XB);             // 4 MB
  float*    x1   = (float*)(ws + (KSPLIT + 1) * XB);       // 4 MB
  _Float16* bhi  = (_Float16*)(ws + (KSPLIT + 2) * XB);    // 2 MB
  _Float16* blo  = bhi + (size_t)NN * HH;                  // 2 MB

  dim3 blk(256);
  dim3 gridPack(NN * HH / 256);
  dim3 gridGemm(NN / 256, KSPLIT);
  dim3 gridRed(NN * HH / 256);
  dim3 gridDense(NN / 4);

  // hop 0
  pack_x_f16<<<gridPack, blk, 0, stream>>>(x0, bhi, blo);
  gcn_agg_wmma<<<gridGemm, blk, 0, stream>>>(adj, bhi, blo, part);
  reduce_parts<<<gridRed, blk, 0, stream>>>(part, agg);
  dense_tanh<<<gridDense, blk, 0, stream>>>(agg, x0, W, x1);
  // hop 1
  pack_x_f16<<<gridPack, blk, 0, stream>>>(x1, bhi, blo);
  gcn_agg_wmma<<<gridGemm, blk, 0, stream>>>(adj, bhi, blo, part);
  reduce_parts<<<gridRed, blk, 0, stream>>>(part, agg);
  dense_tanh<<<gridDense, blk, 0, stream>>>(agg, x1, W + 128 * 64, out);
}
